// LLM_63677185130747
// MI455X (gfx1250) — matrix operand order repaired and checked by reading, not compiled, not run on hardware
//
#include <hip/hip_runtime.h>

// ---------------------------------------------------------------------------
// LLM forward for gfx1250 (MI455X): bf16 WMMA GEMMs (f32 accumulate) with
// Tensor-Data-Mover (TDM) double-buffered LDS tile staging. All GEMMs take
// B as [N,K] (BTRANS) so every tile is a plain 2D TDM descriptor.
// Block tile 128(M) x 256(N) x 32(K); 8 waves; wave tile 64x64 = 16 WMMA/step.
// ---------------------------------------------------------------------------

typedef __bf16 bf16_t;
typedef __bf16 v16bf __attribute__((ext_vector_type(16)));
typedef float  v8f   __attribute__((ext_vector_type(8)));
typedef unsigned int v4u __attribute__((ext_vector_type(4)));
typedef int v4i __attribute__((ext_vector_type(4)));
typedef int v8i __attribute__((ext_vector_type(8)));

#define D_MODEL  1024
#define D_HIDDEN 4096
#define N_BLOCKS 4
#define D_VOCAB  32000
#define BATCH    2
#define SEQ      2048
#define M_ALL    (BATCH * SEQ)   // 4096 token rows

// -------------------------------- reductions -------------------------------
__device__ __forceinline__ float wave_sum(float v) {
#pragma unroll
  for (int m = 16; m > 0; m >>= 1) v += __shfl_xor(v, m, 32);
  return v;
}
__device__ __forceinline__ float wave_max(float v) {
#pragma unroll
  for (int m = 16; m > 0; m >>= 1) v = fmaxf(v, __shfl_xor(v, m, 32));
  return v;
}

// ------------------------- TDM tile load (rows x 32 bf16) ------------------
// Issues TENSOR_LOAD_TO_LDS for a rows x 32-col bf16 tile whose rows are
// ld_elems apart in global memory. LDS destination gets 1 DWORD of padding
// after every 16 DWORDs (64B row) -> row stride 34 bf16, matching [..][34].
__device__ __forceinline__ void tdm_load_tile_32w(void* lds_dst,
                                                  const bf16_t* gsrc,
                                                  unsigned ld_elems,
                                                  unsigned rows) {
  const unsigned lds_addr = (unsigned)(size_t)lds_dst;      // low 32b = LDS offset
  const unsigned long long ga = (unsigned long long)(size_t)gsrc;
  v4u g0;
  g0[0] = 1u;                                               // count=1 (valid)
  g0[1] = lds_addr;                                         // lds_addr [63:32]
  g0[2] = (unsigned)ga;                                     // global_addr [95:64]
  g0[3] = (unsigned)((ga >> 32) & 0x01ffffffull)            // global_addr [120:96]
          | (2u << 30);                                     // type=2 ("image")
  v8i g1;
  g1[0] = (int)((1u << 16)      // data_size = 1 -> 2 bytes
              | (1u << 20)      // pad_enable
              | (3u << 22));    // pad_interval = 3 -> every 16 DWORDs, amount 1 DW
  g1[1] = (int)(32u << 16);     // tensor_dim0 = 32 (lo16 at bits 63:48)
  g1[2] = (int)(rows << 16);    // tensor_dim0 hi = 0 | tensor_dim1 = rows (lo16)
  g1[3] = (int)(32u << 16);     // tensor_dim1 hi = 0 | tile_dim0 = 32
  g1[4] = (int)rows;            // tile_dim1 = rows | tile_dim2 = 0
  g1[5] = (int)ld_elems;        // tensor_dim0_stride lo32 (elements)
  g1[6] = 0;                    // stride hi16 | tensor_dim1_stride lo16
  g1[7] = 0;
  v4i z4 = {0, 0, 0, 0};
#if defined(__clang_major__) && (__clang_major__ >= 23)
  v8i z8 = {0, 0, 0, 0, 0, 0, 0, 0};
  __builtin_amdgcn_tensor_load_to_lds(g0, g1, z4, z4, z8, 0);
#else
  __builtin_amdgcn_tensor_load_to_lds(g0, g1, z4, z4, 0);
#endif
}

// -------------------------------- embed gather -----------------------------
__global__ __launch_bounds__(256) void k_embed(const int* __restrict__ tokens,
                                               const float* __restrict__ emb,
                                               float* __restrict__ x) {
  const int row = blockIdx.x;
  const int tok = tokens[row];
  const float* e = emb + (size_t)tok * D_MODEL;
  float* xr = x + (size_t)row * D_MODEL;
#pragma unroll
  for (int i = 0; i < 4; ++i) {
    const int c = threadIdx.x + i * 256;
    xr[c] = e[c];
  }
}

// -------------------------------- f32 -> bf16 ------------------------------
__global__ __launch_bounds__(256) void k_f32_to_bf16(const float* __restrict__ src,
                                                     bf16_t* __restrict__ dst,
                                                     long n) {
  long i = (long)blockIdx.x * 256 + threadIdx.x;
  const long stride = (long)gridDim.x * 256;
  for (; i < n; i += stride) dst[i] = (bf16_t)src[i];
}

// --------------------- f32 [K][N] -> bf16 [N][K] transpose -----------------
__global__ __launch_bounds__(256) void k_transpose_bf16(const float* __restrict__ src,
                                                        bf16_t* __restrict__ dst,
                                                        int K, int N) {
  __shared__ float t[32][33];
  const int n0 = blockIdx.x * 32;
  const int k0 = blockIdx.y * 32;
  const int tx = threadIdx.x & 31;
  const int ty = threadIdx.x >> 5;          // 0..7
#pragma unroll
  for (int i = 0; i < 4; ++i) {
    const int kk = ty + i * 8;
    t[kk][tx] = src[(size_t)(k0 + kk) * N + (n0 + tx)];
  }
  __syncthreads();
#pragma unroll
  for (int i = 0; i < 4; ++i) {
    const int nn = ty + i * 8;
    dst[(size_t)(n0 + nn) * K + (k0 + tx)] = (bf16_t)t[tx][nn];
  }
}

// -------------------------------- LayerNorm --------------------------------
__global__ __launch_bounds__(256) void k_layernorm(const float* __restrict__ x,
                                                   const float* __restrict__ g,
                                                   const float* __restrict__ beta,
                                                   bf16_t* __restrict__ out) {
  __shared__ float red[8];
  const int row = blockIdx.x;
  const int tid = threadIdx.x;
  const float* xr = x + (size_t)row * D_MODEL;
  float v[4];
  float s = 0.f;
#pragma unroll
  for (int i = 0; i < 4; ++i) { v[i] = xr[tid + i * 256]; s += v[i]; }
  s = wave_sum(s);
  if ((tid & 31) == 0) red[tid >> 5] = s;
  __syncthreads();
  float tot = 0.f;
#pragma unroll
  for (int i = 0; i < 8; ++i) tot += red[i];
  const float mean = tot * (1.f / D_MODEL);
  __syncthreads();
  float var = 0.f;
#pragma unroll
  for (int i = 0; i < 4; ++i) { const float d = v[i] - mean; var += d * d; }
  var = wave_sum(var);
  if ((tid & 31) == 0) red[tid >> 5] = var;
  __syncthreads();
  float vtot = 0.f;
#pragma unroll
  for (int i = 0; i < 8; ++i) vtot += red[i];
  const float rstd = rsqrtf(vtot * (1.f / D_MODEL) + 1e-5f);
  bf16_t* orow = out + (size_t)row * D_MODEL;
#pragma unroll
  for (int i = 0; i < 4; ++i) {
    const int c = tid + i * 256;
    orow[c] = (bf16_t)((v[i] - mean) * rstd * g[c] + beta[c]);
  }
}

// -------------------------------- softmax (causal) -------------------------
__global__ __launch_bounds__(256) void k_softmax(const float* __restrict__ scores,
                                                 bf16_t* __restrict__ attn) {
  __shared__ float red[8];
  const int row = blockIdx.x;                 // b*SEQ + s
  const int s = row & (SEQ - 1);
  const int tid = threadIdx.x;
  const float* sr = scores + (size_t)row * SEQ;
  bf16_t* ar = attn + (size_t)row * SEQ;
  const float inv_scale = 0.03125f;           // 1/sqrt(1024)

  float v[8];
  float vmax = -1e30f;
#pragma unroll
  for (int i = 0; i < 8; ++i) {
    const int t = tid + i * 256;
    const float val = (t <= s) ? sr[t] * inv_scale : -1e30f;
    v[i] = val;
    vmax = fmaxf(vmax, val);
  }
  vmax = wave_max(vmax);
  if ((tid & 31) == 0) red[tid >> 5] = vmax;
  __syncthreads();
  float m = red[0];
#pragma unroll
  for (int i = 1; i < 8; ++i) m = fmaxf(m, red[i]);
  __syncthreads();
  float ssum = 0.f;
#pragma unroll
  for (int i = 0; i < 8; ++i) {
    const int t = tid + i * 256;
    const float e = (t <= s) ? __expf(v[i] - m) : 0.f;
    v[i] = e;
    ssum += e;
  }
  ssum = wave_sum(ssum);
  if ((tid & 31) == 0) red[tid >> 5] = ssum;
  __syncthreads();
  float tot = 0.f;
#pragma unroll
  for (int i = 0; i < 8; ++i) tot += red[i];
  const float rinv = 1.f / tot;
#pragma unroll
  for (int i = 0; i < 8; ++i) ar[tid + i * 256] = (bf16_t)(v[i] * rinv);
}

// -------------------------------- WMMA GEMM --------------------------------
// C[M,N] = epilogue( A[M,K] @ B^T ), with B stored [N,K] (row stride ldb).
// Block tile 128x256x32, 8 waves arranged 2(M) x 4(N); wave tile 64x64 =
// 4x4 16x16 fragments -> 16 WMMA per K-step. TDM double-buffered staging:
// wave 0 issues the A-tile DMA, wave 1 the B-tile DMA (scalar-branched via
// readfirstlane so skipped waves really branch around the EXEC-ignoring TDM
// ops); everyone executes s_wait_tensorcnt(0) (no-op for non-issuing waves),
// then one barrier per K-step releases the workgroup.
template <typename OutT, bool BIAS, bool ACT, bool RESID, bool OTRANS>
__global__ __launch_bounds__(256) void k_gemm(const bf16_t* __restrict__ A,
                                              const bf16_t* __restrict__ B,
                                              const float* __restrict__ bias,
                                              OutT* __restrict__ C,
                                              int M, int N, int K, int ldb) {
  __shared__ bf16_t As[2][128][34];
  __shared__ bf16_t Bs[2][256][34];

  const int tid  = threadIdx.x;
  const int lane = tid & 31;
  const int lh   = lane >> 4;
  const int ln16 = lane & 15;
  // wave id as an SGPR so the TDM-issue guards become scalar branches
  const int wid_u = __builtin_amdgcn_readfirstlane(tid) >> 5;
  const int waveM = wid_u >> 2;   // 0..1 -> 64-row strip
  const int waveN = wid_u & 3;    // 0..3 -> 64-col strip

  const int m0 = blockIdx.y * 128;
  const int n0 = blockIdx.x * 256;

  const bf16_t* Abase = A + (size_t)m0 * K;
  const bf16_t* Bbase = B + (size_t)n0 * ldb;
  const int nk = K >> 5;

  if (wid_u == 0) {
    tdm_load_tile_32w(&As[0][0][0], Abase, (unsigned)K, 128u);
  } else if (wid_u == 1) {
    tdm_load_tile_32w(&Bs[0][0][0], Bbase, (unsigned)ldb, 256u);
  }

  v8f acc[4][4];
#pragma unroll
  for (int i = 0; i < 4; ++i)
#pragma unroll
    for (int j = 0; j < 4; ++j)
#pragma unroll
      for (int e = 0; e < 8; ++e) acc[i][j][e] = 0.f;

  for (int kt = 0; kt < nk; ++kt) {
    const int cur = kt & 1;
    __builtin_amdgcn_s_wait_tensorcnt(0);   // own counter; 0 for non-issuers
    __syncthreads();                        // buf[cur] ready; buf[cur^1] free
    if ((kt + 1) < nk) {
      if (wid_u == 0) {
        tdm_load_tile_32w(&As[cur ^ 1][0][0], Abase + (size_t)(kt + 1) * 32,
                          (unsigned)K, 128u);
      } else if (wid_u == 1) {
        tdm_load_tile_32w(&Bs[cur ^ 1][0][0], Bbase + (size_t)(kt + 1) * 32,
                          (unsigned)ldb, 256u);
      }
    }

    // ---- load all fragments first (distinct regs -> fewer waits/NOPs) ----
    v16bf af[4];
#pragma unroll
    for (int mf = 0; mf < 4; ++mf) {
      const int row = waveM * 64 + mf * 16 + ln16;
#pragma unroll
      for (int vv = 0; vv < 8; ++vv) {
        // A 16x32 layout: half0 VGPR v -> K={2v,2v+1} (v<4) / {16+2(v-4),..};
        // half1 adds +8.
        const int kb = (vv < 4 ? 2 * vv : 16 + 2 * (vv - 4)) + 8 * lh;
        af[mf][2 * vv]     = As[cur][row][kb];
        af[mf][2 * vv + 1] = As[cur][row][kb + 1];
      }
    }
    v16bf bfrag[4];
#pragma unroll
    for (int nf = 0; nf < 4; ++nf) {
      const int col = waveN * 64 + nf * 16 + ln16;
#pragma unroll
      for (int vv = 0; vv < 8; ++vv) {
        // B 32x16 layout: VGPR v -> K={2v,2v+1} (v<4) / {16+2(v-4),..}, half1 adds +8; N = lane&15.
        const int kb = (vv < 4 ? 2 * vv : 16 + 2 * (vv - 4)) + 8 * lh;
        bfrag[nf][2 * vv]     = Bs[cur][col][kb];
        bfrag[nf][2 * vv + 1] = Bs[cur][col][kb + 1];
      }
    }
    // ---- 16 WMMAs ----
#pragma unroll
    for (int nf = 0; nf < 4; ++nf)
#pragma unroll
      for (int mf = 0; mf < 4; ++mf)
        acc[mf][nf] = __builtin_amdgcn_wmma_f32_16x16x32_bf16(
            false, af[mf], false, bfrag[nf], (short)0, acc[mf][nf], false, false);
  }

  // Epilogue: C layout per ISA (lane&15 = col, row = vgpr + 8*(lane>>4)).
#pragma unroll
  for (int mf = 0; mf < 4; ++mf) {
#pragma unroll
    for (int nf = 0; nf < 4; ++nf) {
      const int col = n0 + waveN * 64 + nf * 16 + ln16;
      const float bv = BIAS ? bias[col] : 0.f;
#pragma unroll
      for (int vv = 0; vv < 8; ++vv) {
        const int row = m0 + waveM * 64 + mf * 16 + vv + 8 * lh;
        float val = acc[mf][nf][vv] + bv;
        if (ACT) val = (val > 0.f) ? val : 0.01f * val;   // LeakyReLU
        const size_t o = OTRANS ? ((size_t)col * M + row)
                                : ((size_t)row * N + col);
        if (RESID) val += (float)C[o];
        C[o] = (OutT)val;
      }
    }
  }
}

// ------------------------------- host driver -------------------------------
extern "C" void kernel_launch(void* const* d_in, const int* in_sizes, int n_in,
                              void* d_out, int out_size, void* d_ws, size_t ws_size,
                              hipStream_t stream) {
  (void)in_sizes; (void)n_in; (void)out_size; (void)ws_size;

  const int*   tokens = (const int*)d_in[0];
  const float* emb    = (const float*)d_in[1];
  const float* qk     = (const float*)d_in[2];
  const float* ov     = (const float*)d_in[3];
  const float* w_up   = (const float*)d_in[4];
  const float* b_up   = (const float*)d_in[5];
  const float* w_hid  = (const float*)d_in[6];
  const float* b_hid  = (const float*)d_in[7];
  const float* w_down = (const float*)d_in[8];
  const float* b_down = (const float*)d_in[9];
  const float* ln1_g  = (const float*)d_in[10];
  const float* ln1_b  = (const float*)d_in[11];
  const float* ln2_g  = (const float*)d_in[12];
  const float* ln2_b  = (const float*)d_in[13];
  float* out = (float*)d_out;

  char* ws = (char*)d_ws;
  size_t off = 0;
  auto take = [&](size_t bytes) -> void* {
    void* p = (void*)(ws + off);
    off += (bytes + 255) & ~(size_t)255;
    return p;
  };
  float*  x      = (float*) take((size_t)M_ALL * D_MODEL * 4);
  bf16_t* h_bf   = (bf16_t*)take((size_t)M_ALL * D_MODEL * 2);
  bf16_t* q_bf   = (bf16_t*)take((size_t)M_ALL * D_MODEL * 2);
  bf16_t* hvT    = (bf16_t*)take((size_t)D_MODEL * M_ALL * 2); // [D][M] transposed
  float*  scores = (float*) take((size_t)BATCH * SEQ * SEQ * 4);
  bf16_t* attn   = (bf16_t*)take((size_t)BATCH * SEQ * SEQ * 2);
  bf16_t* m_bf   = (bf16_t*)take((size_t)M_ALL * D_MODEL * 2);
  bf16_t* t1_bf  = (bf16_t*)take((size_t)M_ALL * D_HIDDEN * 2);
  bf16_t* t2_bf  = (bf16_t*)take((size_t)M_ALL * D_HIDDEN * 2);
  bf16_t* wT     = (bf16_t*)take((size_t)D_HIDDEN * D_HIDDEN * 2); // [N][K]
  bf16_t* emb_bf = (bf16_t*)take((size_t)D_VOCAB * D_MODEL * 2);   // [V][D]=[N][K]
  bf16_t* x_bf   = (bf16_t*)take((size_t)M_ALL * D_MODEL * 2);

  k_embed<<<M_ALL, 256, 0, stream>>>(tokens, emb, x);
  k_f32_to_bf16<<<4096, 256, 0, stream>>>(emb, emb_bf, (long)D_VOCAB * D_MODEL);

  for (int l = 0; l < N_BLOCKS; ++l) {
    const float* qk_l     = qk     + (size_t)l * D_MODEL * D_MODEL;
    const float* ov_l     = ov     + (size_t)l * D_MODEL * D_MODEL;
    const float* w_up_l   = w_up   + (size_t)l * D_MODEL * D_HIDDEN;
    const float* b_up_l   = b_up   + (size_t)l * D_HIDDEN;
    const float* w_hid_l  = w_hid  + (size_t)l * D_HIDDEN * D_HIDDEN;
    const float* b_hid_l  = b_hid  + (size_t)l * D_HIDDEN;
    const float* w_down_l = w_down + (size_t)l * D_HIDDEN * D_MODEL;
    const float* b_down_l = b_down + (size_t)l * D_MODEL;

    // ---- attention ----
    k_layernorm<<<M_ALL, 256, 0, stream>>>(x, ln1_g + l * D_MODEL,
                                           ln1_b + l * D_MODEL, h_bf);
    // q = h @ qk           (B = qk^T [1024][1024])
    k_transpose_bf16<<<dim3(D_MODEL / 32, D_MODEL / 32), 256, 0, stream>>>(
        qk_l, wT, D_MODEL, D_MODEL);
    k_gemm<bf16_t, false, false, false, false>
        <<<dim3(D_MODEL / 256, M_ALL / 128), 256, 0, stream>>>(
            h_bf, wT, nullptr, q_bf, M_ALL, D_MODEL, D_MODEL, D_MODEL);
    // scores_b = q_b @ h_b^T   (B = h_b [2048][1024], already [N][K])
    for (int b = 0; b < BATCH; ++b) {
      k_gemm<float, false, false, false, false>
          <<<dim3(SEQ / 256, SEQ / 128), 256, 0, stream>>>(
              q_bf + (size_t)b * SEQ * D_MODEL,
              h_bf + (size_t)b * SEQ * D_MODEL, nullptr,
              scores + (size_t)b * SEQ * SEQ, SEQ, SEQ, D_MODEL, D_MODEL);
    }
    k_softmax<<<BATCH * SEQ, 256, 0, stream>>>(scores, attn);
    // hvT = (h @ ov)^T     (B = ov^T; epilogue writes transposed [D][M])
    k_transpose_bf16<<<dim3(D_MODEL / 32, D_MODEL / 32), 256, 0, stream>>>(
        ov_l, wT, D_MODEL, D_MODEL);
    k_gemm<bf16_t, false, false, false, true>
        <<<dim3(D_MODEL / 256, M_ALL / 128), 256, 0, stream>>>(
            h_bf, wT, nullptr, hvT, M_ALL, D_MODEL, D_MODEL, D_MODEL);
    // x += attn_b @ hv_b   (B = hvT[:, b*SEQ:(b+1)*SEQ] as [1024][2048], ldb=M_ALL)
    for (int b = 0; b < BATCH; ++b) {
      k_gemm<float, false, false, true, false>
          <<<dim3(D_MODEL / 256, SEQ / 128), 256, 0, stream>>>(
              attn + (size_t)b * SEQ * SEQ, hvT + (size_t)b * SEQ, nullptr,
              x + (size_t)b * SEQ * D_MODEL, SEQ, D_MODEL, SEQ, M_ALL);
    }

    // ---- MLP ----
    k_layernorm<<<M_ALL, 256, 0, stream>>>(x, ln2_g + l * D_MODEL,
                                           ln2_b + l * D_MODEL, m_bf);
    k_transpose_bf16<<<dim3(D_HIDDEN / 32, D_MODEL / 32), 256, 0, stream>>>(
        w_up_l, wT, D_MODEL, D_HIDDEN);          // [1024][4096] -> [4096][1024]
    k_gemm<bf16_t, true, true, false, false>
        <<<dim3(D_HIDDEN / 256, M_ALL / 128), 256, 0, stream>>>(
            m_bf, wT, b_up_l, t1_bf, M_ALL, D_HIDDEN, D_MODEL, D_MODEL);
    k_transpose_bf16<<<dim3(D_HIDDEN / 32, D_HIDDEN / 32), 256, 0, stream>>>(
        w_hid_l, wT, D_HIDDEN, D_HIDDEN);
    k_gemm<bf16_t, true, true, false, false>
        <<<dim3(D_HIDDEN / 256, M_ALL / 128), 256, 0, stream>>>(
            t1_bf, wT, b_hid_l, t2_bf, M_ALL, D_HIDDEN, D_HIDDEN, D_HIDDEN);
    k_transpose_bf16<<<dim3(D_MODEL / 32, D_HIDDEN / 32), 256, 0, stream>>>(
        w_down_l, wT, D_HIDDEN, D_MODEL);        // [4096][1024] -> [1024][4096]
    k_gemm<float, true, true, true, false>
        <<<dim3(D_MODEL / 256, M_ALL / 128), 256, 0, stream>>>(
            t2_bf, wT, b_down_l, x, M_ALL, D_MODEL, D_HIDDEN, D_HIDDEN);
  }

  // ---- logits = x @ emb^T   (B = emb_bf [32000][1024], already [N][K]) ----
  k_f32_to_bf16<<<2048, 256, 0, stream>>>(x, x_bf, (long)M_ALL * D_MODEL);
  k_gemm<float, false, false, false, false>
      <<<dim3(D_VOCAB / 256, M_ALL / 128), 256, 0, stream>>>(
          x_bf, emb_bf, nullptr, out, M_ALL, D_VOCAB, D_MODEL, D_MODEL);
}
